// TimeAwareNodeModel_57200374448179
// MI455X (gfx1250) — matrix operand order repaired and checked by reading, not compiled, not run on hardware
//
#include <hip/hip_runtime.h>
#include <hip/hip_bf16.h>

// ---------------------------------------------------------------------------
// TimeAwareNodeModel fused bf16-WMMA implementation for gfx1250 (MI455X)
//
//   edge_in[E,192] = concat(x[n2], edge_attr)
//   fut  = relu(edge_in@W1f+b1f)@W2f+b2f, masked n1<n2, scatter-sum -> n1
//   past = relu(edge_in@W1p+b1p)@W2p+b2p, masked n1>n2, scatter-mean -> n1
//   out  = relu([past‖fut]@W1c+b1c)@W2c+b2c + x
//
// All GEMMs use v_wmma_f32_16x16x32_bf16 with f32 accumulation.
// Edge kernel uses M=32 per wave (two A row-tiles per B fragment) to halve
// weight-fragment bandwidth per WMMA.
// ---------------------------------------------------------------------------

typedef __attribute__((ext_vector_type(16))) __bf16 v16bf;
typedef __attribute__((ext_vector_type(8)))  __bf16 v8bf;
typedef __attribute__((ext_vector_type(8)))  float  v8f;

#define NN   100000
#define NE   600000
#define DX   128
#define DE   64
#define HPAD 264   // 256 + 8 bf16 pad per row (LDS bank shift)

__device__ __forceinline__ v8f wmma_bf16(v16bf a, v16bf b, v8f c) {
    return __builtin_amdgcn_wmma_f32_16x16x32_bf16(
        false, a, false, b, (short)0, c, false, false);
}

// Load one B fragment from packed weights: layout [kTile][nTile][lane][16]
__device__ __forceinline__ v16bf load_bfrag(const __bf16* __restrict__ p,
                                            int kT, int nT, int nTiles, int lane) {
    return *(const v16bf*)(p + ((((size_t)kT * nTiles + nT) * 32 + lane) << 4));
}

// Build a bf16 A fragment (16x32) from a row-major f32 row pointer at k-offset 0.
// Lane layout (ISA 7.12.2): lane = hi*16+m holds K in {hi*8..hi*8+7, 16+hi*8..}
__device__ __forceinline__ v16bf make_afrag_f32(const float* __restrict__ src,
                                                int hi, float scale) {
    v8f c1 = *(const v8f*)(src + hi * 8);
    v8f c2 = *(const v8f*)(src + 16 + hi * 8);
    v16bf av;
#pragma unroll
    for (int j = 0; j < 8; ++j) {
        av[j]     = (__bf16)(c1[j] * scale);
        av[8 + j] = (__bf16)(c2[j] * scale);
    }
    return av;
}

// Build a bf16 A fragment from an LDS bf16 row (HPAD stride) at k-offset kt*32
__device__ __forceinline__ v16bf make_afrag_lds(const __bf16* row, int hi) {
    v8bf h1 = *(const v8bf*)(row + hi * 8);
    v8bf h2 = *(const v8bf*)(row + 16 + hi * 8);
    v16bf av;
#pragma unroll
    for (int j = 0; j < 8; ++j) { av[j] = h1[j]; av[8 + j] = h2[j]; }
    return av;
}

// ---------------------------------------------------------------------------
// Workspace zeroing (accumulators must be 0 each call)
// ---------------------------------------------------------------------------
__global__ void zero_f32_kernel(float* __restrict__ p, long long n) {
    long long i = (long long)blockIdx.x * blockDim.x + threadIdx.x;
    long long stride = (long long)gridDim.x * blockDim.x;
    for (; i < n; i += stride) p[i] = 0.0f;
}

// ---------------------------------------------------------------------------
// Pack a f32 [K][N] weight into bf16 B-fragment layout [K/32][N/16][32][16]
//   lane: nlo = lane&15 (column), khalf = lane>>4; element j -> K = kT*32+khalf*8+(j&7)+(j>>3)*16
// ---------------------------------------------------------------------------
__global__ void pack_weight_kernel(const float* __restrict__ src,
                                   __bf16* __restrict__ dst, int K, int N) {
    int tid = blockIdx.x * blockDim.x + threadIdx.x;
    if (tid >= K * N) return;
    int j     = tid & 15;
    int lane  = (tid >> 4) & 31;
    int rest  = tid >> 9;
    int nTiles = N >> 4;
    int nT    = rest % nTiles;
    int kT    = rest / nTiles;
    int khalf = lane >> 4;
    int nlo   = lane & 15;
    int kIdx  = kT * 32 + khalf * 8 + (j & 7) + (j >> 3) * 16;
    int nIdx  = nT * 16 + nlo;
    dst[tid] = (__bf16)src[(size_t)kIdx * N + nIdx];
}

// ---------------------------------------------------------------------------
// Past-edge counts per destination node (for scatter-mean)
// ---------------------------------------------------------------------------
__global__ void cnt_kernel(const int* __restrict__ eidx, float* __restrict__ cnt) {
    int e = blockIdx.x * blockDim.x + threadIdx.x;
    if (e >= NE) return;
    int d1 = eidx[e], d2 = eidx[NE + e];
    if (d1 > d2) atomicAdd(&cnt[d1], 1.0f);
}

// ---------------------------------------------------------------------------
// Fused edge kernel: 2 waves/block, one 32-edge tile per wave, both branches.
// Two A row-tiles share each B fragment -> 2 WMMA per B load.
// ---------------------------------------------------------------------------
__global__ void __launch_bounds__(64)
edge_mlp_kernel(const float* __restrict__ x,
                const int*   __restrict__ eidx,
                const float* __restrict__ eattr,
                const __bf16* __restrict__ pW1f, const float* __restrict__ b1f,
                const __bf16* __restrict__ pW2f, const float* __restrict__ b2f,
                const __bf16* __restrict__ pW1p, const float* __restrict__ b1p,
                const __bf16* __restrict__ pW2p, const float* __restrict__ b2p,
                float* __restrict__ futAgg, float* __restrict__ pastSum)
{
    __shared__ __bf16 hid[2][32 * HPAD];
    const int lane  = threadIdx.x & 31;
    const int w     = threadIdx.x >> 5;
    const int hi    = lane >> 4;
    const int nlo   = lane & 15;
    const int eBase = (blockIdx.x * 2 + w) * 32;

    // ---- endpoints + masks for the scatter (C layout: M = r + 8*hi) ----
    int  dstr[2][8];
    bool mf[2][8], mp[2][8];
#pragma unroll
    for (int t = 0; t < 2; ++t) {
#pragma unroll
        for (int r = 0; r < 8; ++r) {
            const int e  = eBase + t * 16 + r + 8 * hi;
            const int d1 = eidx[e];
            const int d2 = eidx[NE + e];
            dstr[t][r] = d1; mf[t][r] = (d1 < d2); mp[t][r] = (d1 > d2);
        }
    }

    __bf16* hrow = &hid[w][0];

#pragma unroll 1
    for (int br = 0; br < 2; ++br) {
        const __bf16* pW1 = br ? pW1p : pW1f;
        const __bf16* pW2 = br ? pW2p : pW2f;
        const float*  b1  = br ? b1p  : b1f;
        const float*  b2  = br ? b2p  : b2f;
        float*        agg = br ? pastSum : futAgg;

        // ---- A fragments: 32 edges x 192 (x[n2] ‖ edge_attr) ----
        // Rebuilt per branch to keep GEMM2 register pressure low.
        v16bf a[2][6];
#pragma unroll
        for (int t = 0; t < 2; ++t) {
            const int em   = eBase + t * 16 + nlo;
            const int node = eidx[NE + em];        // n2 (source features)
#pragma unroll
            for (int kt = 0; kt < 6; ++kt) {
                const float* src = (kt < 4)
                    ? (x     + (size_t)node * DX + kt * 32)
                    : (eattr + (size_t)em   * DE + (kt - 4) * 32);
                a[t][kt] = make_afrag_f32(src, hi, 1.0f);
            }
        }

        // GEMM1: hidden[32x256] = relu(A @ W1 + b1)   (K = 192, 6 WMMA steps)
        for (int nT = 0; nT < 16; ++nT) {
            const float bv = b1[nT * 16 + nlo];
            v8f c0, c1;
#pragma unroll
            for (int r = 0; r < 8; ++r) { c0[r] = bv; c1[r] = bv; }
#pragma unroll
            for (int kt = 0; kt < 6; ++kt) {
                v16bf b = load_bfrag(pW1, kt, nT, 16, lane);
                c0 = wmma_bf16(a[0][kt], b, c0);
                c1 = wmma_bf16(a[1][kt], b, c1);
            }
            const int col = nT * 16 + nlo;
#pragma unroll
            for (int r = 0; r < 8; ++r) {
                hrow[(r + 8 * hi) * HPAD + col]      = (__bf16)fmaxf(c0[r], 0.0f);
                hrow[(16 + r + 8 * hi) * HPAD + col] = (__bf16)fmaxf(c1[r], 0.0f);
            }
        }
        __syncthreads();

        // Reload hidden in A-fragment layout (32 x 256, 8 K-tiles x 2 row-tiles)
        v16bf ah[2][8];
#pragma unroll
        for (int t = 0; t < 2; ++t)
#pragma unroll
            for (int kt = 0; kt < 8; ++kt)
                ah[t][kt] = make_afrag_lds(hrow + (t * 16 + nlo) * HPAD + kt * 32, hi);
        __syncthreads();

        // GEMM2: out[32x128] = hidden @ W2 + b2   (K = 256, 8 WMMA steps)
        for (int nT = 0; nT < 8; ++nT) {
            const float bv = b2[nT * 16 + nlo];
            v8f c0, c1;
#pragma unroll
            for (int r = 0; r < 8; ++r) { c0[r] = bv; c1[r] = bv; }
#pragma unroll
            for (int kt = 0; kt < 8; ++kt) {
                v16bf b = load_bfrag(pW2, kt, nT, 8, lane);
                c0 = wmma_bf16(ah[0][kt], b, c0);
                c1 = wmma_bf16(ah[1][kt], b, c1);
            }
            const int col = nT * 16 + nlo;
#pragma unroll
            for (int r = 0; r < 8; ++r) {
                const bool m0 = br ? mp[0][r] : mf[0][r];
                const bool m1 = br ? mp[1][r] : mf[1][r];
                if (m0) atomicAdd(&agg[(size_t)dstr[0][r] * DX + col], c0[r]);
                if (m1) atomicAdd(&agg[(size_t)dstr[1][r] * DX + col], c1[r]);
            }
        }
    }
}

// ---------------------------------------------------------------------------
// Combine kernel: 2 waves/block, 16 nodes per wave.
//   flow = [past_sum/max(cnt,1) ‖ fut], out = relu(flow@W1c+b1c)@W2c+b2c + x
// ---------------------------------------------------------------------------
__global__ void __launch_bounds__(64)
combine_kernel(const float* __restrict__ x,
               const float* __restrict__ futAgg,
               const float* __restrict__ pastSum,
               const float* __restrict__ cnt,
               const __bf16* __restrict__ pW1c, const float* __restrict__ b1c,
               const __bf16* __restrict__ pW2c, const float* __restrict__ b2c,
               float* __restrict__ out)
{
    __shared__ __bf16 hid[2][16 * HPAD];
    const int lane  = threadIdx.x & 31;
    const int w     = threadIdx.x >> 5;
    const int hi    = lane >> 4;
    const int nlo   = lane & 15;
    const int nBase = (blockIdx.x * 2 + w) * 16;

    // A fragments: 16 nodes x 256
    v16bf a[8];
    {
        const int   node = nBase + nlo;
        const float inv  = 1.0f / fmaxf(cnt[node], 1.0f);
#pragma unroll
        for (int kt = 0; kt < 8; ++kt) {
            const bool  past = (kt < 4);
            const float* src = past
                ? (pastSum + (size_t)node * DX + kt * 32)
                : (futAgg  + (size_t)node * DX + (kt - 4) * 32);
            a[kt] = make_afrag_f32(src, hi, past ? inv : 1.0f);
        }
    }

    __bf16* hrow = &hid[w][0];

    // GEMM1: hidden = relu(flow @ W1c + b1c)  (K = 256)
    for (int nT = 0; nT < 16; ++nT) {
        const float bv = b1c[nT * 16 + nlo];
        v8f c;
#pragma unroll
        for (int r = 0; r < 8; ++r) c[r] = bv;
#pragma unroll
        for (int kt = 0; kt < 8; ++kt)
            c = wmma_bf16(a[kt], load_bfrag(pW1c, kt, nT, 16, lane), c);
#pragma unroll
        for (int r = 0; r < 8; ++r)
            hrow[(r + 8 * hi) * HPAD + nT * 16 + nlo] = (__bf16)fmaxf(c[r], 0.0f);
    }
    __syncthreads();

    v16bf ah[8];
#pragma unroll
    for (int kt = 0; kt < 8; ++kt)
        ah[kt] = make_afrag_lds(hrow + nlo * HPAD + kt * 32, hi);

    // GEMM2: out = hidden @ W2c + b2c + x  (K = 256)
    for (int nT = 0; nT < 8; ++nT) {
        const float bv = b2c[nT * 16 + nlo];
        v8f c;
#pragma unroll
        for (int r = 0; r < 8; ++r) c[r] = bv;
#pragma unroll
        for (int kt = 0; kt < 8; ++kt)
            c = wmma_bf16(ah[kt], load_bfrag(pW2c, kt, nT, 8, lane), c);

        const int col = nT * 16 + nlo;
#pragma unroll
        for (int r = 0; r < 8; ++r) {
            const size_t idx = (size_t)(nBase + r + 8 * hi) * DX + col;
            out[idx] = c[r] + x[idx];
        }
    }
}

// ---------------------------------------------------------------------------
// Host-side launch
// ---------------------------------------------------------------------------
static inline size_t align256(size_t v) { return (v + 255) & ~(size_t)255; }

extern "C" void kernel_launch(void* const* d_in, const int* in_sizes, int n_in,
                              void* d_out, int out_size, void* d_ws, size_t ws_size,
                              hipStream_t stream) {
    (void)in_sizes; (void)n_in; (void)out_size; (void)ws_size;

    const float* x     = (const float*)d_in[0];
    const int*   eidx  = (const int*)  d_in[1];
    const float* eattr = (const float*)d_in[2];
    const float* W1f = (const float*)d_in[5],  *b1f = (const float*)d_in[6];
    const float* W2f = (const float*)d_in[7],  *b2f = (const float*)d_in[8];
    const float* W1p = (const float*)d_in[9],  *b1p = (const float*)d_in[10];
    const float* W2p = (const float*)d_in[11], *b2p = (const float*)d_in[12];
    const float* W1c = (const float*)d_in[13], *b1c = (const float*)d_in[14];
    const float* W2c = (const float*)d_in[15], *b2c = (const float*)d_in[16];
    float* out = (float*)d_out;

    // Workspace layout
    char* ws = (char*)d_ws;
    size_t off = 0;
    float* futAgg  = (float*)(ws + off); off += align256((size_t)NN * DX * 4);
    float* pastSum = (float*)(ws + off); off += align256((size_t)NN * DX * 4);
    float* cnt     = (float*)(ws + off); off += align256((size_t)NN * 4);
    __bf16* pW1f = (__bf16*)(ws + off); off += align256((size_t)192 * 256 * 2);
    __bf16* pW2f = (__bf16*)(ws + off); off += align256((size_t)256 * 128 * 2);
    __bf16* pW1p = (__bf16*)(ws + off); off += align256((size_t)192 * 256 * 2);
    __bf16* pW2p = (__bf16*)(ws + off); off += align256((size_t)256 * 128 * 2);
    __bf16* pW1c = (__bf16*)(ws + off); off += align256((size_t)256 * 256 * 2);
    __bf16* pW2c = (__bf16*)(ws + off); off += align256((size_t)256 * 128 * 2);

    // 1) Zero accumulators (futAgg, pastSum, cnt are contiguous)
    {
        long long nz = (long long)NN * DX * 2 + NN;
        zero_f32_kernel<<<2048, 256, 0, stream>>>(futAgg, nz);
    }

    // 2) Pack weights into bf16 B-fragment layout
    {
        const int T = 256;
        pack_weight_kernel<<<(192*256 + T - 1)/T, T, 0, stream>>>(W1f, pW1f, 192, 256);
        pack_weight_kernel<<<(256*128 + T - 1)/T, T, 0, stream>>>(W2f, pW2f, 256, 128);
        pack_weight_kernel<<<(192*256 + T - 1)/T, T, 0, stream>>>(W1p, pW1p, 192, 256);
        pack_weight_kernel<<<(256*128 + T - 1)/T, T, 0, stream>>>(W2p, pW2p, 256, 128);
        pack_weight_kernel<<<(256*256 + T - 1)/T, T, 0, stream>>>(W1c, pW1c, 256, 256);
        pack_weight_kernel<<<(256*128 + T - 1)/T, T, 0, stream>>>(W2c, pW2c, 256, 128);
    }

    // 3) Past-edge counts
    cnt_kernel<<<(NE + 255)/256, 256, 0, stream>>>(eidx, cnt);

    // 4) Fused edge MLPs + masked scatter (600000 edges = 9375 blocks * 2 waves * 32)
    edge_mlp_kernel<<<NE/(2*32), 64, 0, stream>>>(
        x, eidx, eattr,
        pW1f, b1f, pW2f, b2f,
        pW1p, b1p, pW2p, b2p,
        futAgg, pastSum);

    // 5) Combine MLP + residual (100000 nodes = 3125 blocks * 2 waves * 16)
    combine_kernel<<<NN/(2*16), 64, 0, stream>>>(
        x, futAgg, pastSum, cnt,
        pW1c, b1c, pW2c, b2c, out);
}
